// EVCBlock_22557168238841
// MI455X (gfx1250) — matrix operand order repaired and checked by reading, not compiled, not run on hardware
//
#include <hip/hip_runtime.h>
#include <math.h>

// ---------------------------------------------------------------------------
// EVCBlock forward for MI455X (gfx1250, wave32).
// Matmul-shaped work runs on v_wmma_f32_16x16x32_f16 with fp32 accumulation.
// Block tile 64(M)x128(N), K staged 64 deep per barrier; each wave owns a
// 32x32 patch -> 8 WMMA issues per LDS stage. For the streaming (non-trans)
// GEMMs the B tile is fetched by the Tensor Data Mover (tensor_load_to_lds,
// TENSORcnt) instead of VALU staging.
// ---------------------------------------------------------------------------

typedef __attribute__((ext_vector_type(16))) _Float16 v16h;
typedef __attribute__((ext_vector_type(8)))  float    v8f;
typedef __attribute__((ext_vector_type(4)))  unsigned int u32x4;
typedef __attribute__((ext_vector_type(8)))  int      i32x8;
typedef __attribute__((ext_vector_type(4)))  int      i32x4;

#define B_   8
#define C1_  256
#define C2_  256
#define CB_  64
#define KC_  64      // num codes
#define HID_ 1024
#define H_   64
#define W_   64
#define HW_  4096

#define WMMA16(a, b, c) __builtin_amdgcn_wmma_f32_16x16x32_f16(false, a, false, b, (short)0, c, false, false)

// ========================= WMMA GEMM (generic) =============================
// Out[m,n] (ldo) = A[m,k] (lda) * X  (+bias[m]) (+= if accum)
//   TRANSB==0 : X element (k,n) at X[k*ldb + n]   (B tile staged by TDM;
//               callers guarantee the 64x128 tile is fully in-bounds:
//               N == ldb, n0+128 <= N, Kd multiple of 64)
//   TRANSB==1 : X element (k,n) at X[n*ldb + k]   (manual staging)
// blockIdx.z applies per-batch strides sA/sX/sO.
// 256 threads = 8 waves in 2(M) x 4(N); wave owns 32x32 of a 64x128 tile.
template<int TRANSB>
__global__ __launch_bounds__(256)
void gemm_wmma(const float* __restrict__ A, long sA, int lda,
               const float* __restrict__ X, long sX, int ldb,
               float* __restrict__ O, long sO, int ldo,
               const float* __restrict__ bias,
               int M, int N, int Kd, int accum)
{
    A += (long)blockIdx.z * sA;
    X += (long)blockIdx.z * sX;
    O += (long)blockIdx.z * sO;
    const int m0 = blockIdx.y * 64, n0 = blockIdx.x * 128;

    __shared__ _Float16 As[64][72];                    // 64 x 64 K (+pad), f16
    __shared__ __align__(16) unsigned char smemB[64 * 128 * 4];
    float    (*Bf)[128] = (float   (*)[128])smemB;     // TDM target (fp32)
    _Float16 (*Bs)[136] = (_Float16(*)[136])smemB;     // manual path (f16)

    const int t = threadIdx.x, lane = t & 31, wave = t >> 5;
    const int wm = wave >> 2, wn = wave & 3;
    const int lh = lane >> 4, l16 = lane & 15;

    v8f acc00 = {}, acc01 = {}, acc10 = {}, acc11 = {};

    for (int k0 = 0; k0 < Kd; k0 += 64) {
        // --- stage A tile 64x64 (fp32 -> f16), float4 loads ---
        #pragma unroll
        for (int i = 0; i < 4; ++i) {
            int e = i * 256 + t, r = e >> 4, c = (e & 15) * 4;
            int gm = m0 + r, gk = k0 + c;
            if (gm < M && gk + 3 < Kd) {
                float4 v = *(const float4*)&A[(long)gm * lda + gk];
                As[r][c + 0] = (_Float16)v.x;
                As[r][c + 1] = (_Float16)v.y;
                As[r][c + 2] = (_Float16)v.z;
                As[r][c + 3] = (_Float16)v.w;
            } else {
                #pragma unroll
                for (int q = 0; q < 4; ++q) {
                    float v = (gm < M && gk + q < Kd) ? A[(long)gm * lda + gk + q] : 0.f;
                    As[r][c + q] = (_Float16)v;
                }
            }
        }
        // --- stage B tile 64(K) x 128(N) ---
        if (TRANSB) {
            #pragma unroll
            for (int i = 0; i < 8; ++i) {
                int e = i * 256 + t;
                int n = e >> 4, c = (e & 15) * 4;      // n: 0..127, k: c..c+3
                int gn = n0 + n, gk = k0 + c;
                if (gn < N && gk + 3 < Kd) {
                    float4 v = *(const float4*)&X[(long)gn * ldb + gk];
                    Bs[c + 0][n] = (_Float16)v.x;
                    Bs[c + 1][n] = (_Float16)v.y;
                    Bs[c + 2][n] = (_Float16)v.z;
                    Bs[c + 3][n] = (_Float16)v.w;
                } else {
                    #pragma unroll
                    for (int q = 0; q < 4; ++q) {
                        float v = (gn < N && gk + q < Kd) ? X[(long)gn * ldb + gk + q] : 0.f;
                        Bs[c + q][n] = (_Float16)v;
                    }
                }
            }
        } else if (wave == 0) {
            // Tensor Data Mover: DMA the 2D tile X[k0..k0+63][n0..n0+127]
            // (row stride ldb, 4-byte elements) into LDS. ISA D# packing
            // (CDNA5 ISA #8.3/#8.4), groups 2/3 zero (2D).
            unsigned long long ga =
                (unsigned long long)(const void*)(X + (long)k0 * ldb + n0);
            unsigned lds = (unsigned)(unsigned long long)(void*)&Bf[0][0];
            unsigned td0 = (unsigned)ldb;              // tensor_dim0
            unsigned td1 = (unsigned)(Kd - k0);        // tensor_dim1 (rows left)
            u32x4 g0;
            g0[0] = 1u;                                 // count=1, user mode
            g0[1] = lds;                                // lds_addr
            g0[2] = (unsigned)(ga & 0xffffffffu);       // global_addr[31:0]
            g0[3] = (unsigned)((ga >> 32) & 0x1ffffffu) // global_addr[56:32]
                  | (2u << 30);                         // type=2 (image)
            i32x8 g1;
            g1[0] = (int)(2u << 16);                    // data_size=4B; no pad
            g1[1] = (int)((td0 & 0xffffu) << 16);       // tensor_dim0[15:0]
            g1[2] = (int)((td0 >> 16) | ((td1 & 0xffffu) << 16));
            g1[3] = (int)((td1 >> 16) | (128u << 16));  // tile_dim0=128
            g1[4] = 64;                                 // tile_dim1=64, tile_dim2=0
            g1[5] = (int)td0;                           // tensor_dim0_stride[31:0]
            g1[6] = 0;                                  // stride hi / dim1_stride
            g1[7] = 0;
            i32x4 gz4 = {0, 0, 0, 0};
            i32x8 gz8 = {0, 0, 0, 0, 0, 0, 0, 0};
            __builtin_amdgcn_tensor_load_to_lds(g0, g1, gz4, gz4, gz8, 0);
            __builtin_amdgcn_s_wait_tensorcnt(0);
        }
        if (!TRANSB && (k0 + 64) < Kd)
            __builtin_prefetch(&A[(long)(m0 + (t & 63)) * lda + k0 + 64], 0, 1);
        __syncthreads();

        // --- 2 K-chunks of 32, 8 WMMAs per stage ---
        #pragma unroll
        for (int kk = 0; kk < 64; kk += 32) {
            v16h a0, a1, b0, b1;
            #pragma unroll
            for (int j = 0; j < 8; ++j) {     // A(16x32) lane layout
                a0[j]     = As[wm * 32 + l16][kk + lh * 8 + j];
                a0[8 + j] = As[wm * 32 + l16][kk + 16 + lh * 8 + j];
                a1[j]     = As[wm * 32 + 16 + l16][kk + lh * 8 + j];
                a1[8 + j] = As[wm * 32 + 16 + l16][kk + 16 + lh * 8 + j];
            }
            #pragma unroll
            for (int j = 0; j < 16; ++j) {    // B(32x16) lane layout
                int kl = kk + (j < 8 ? lh * 8 + j : 16 + lh * 8 + (j - 8));
                if (TRANSB) {
                    b0[j] = Bs[kl][wn * 32 + l16];
                    b1[j] = Bs[kl][wn * 32 + 16 + l16];
                } else {
                    b0[j] = (_Float16)Bf[kl][wn * 32 + l16];
                    b1[j] = (_Float16)Bf[kl][wn * 32 + 16 + l16];
                }
            }
            acc00 = WMMA16(a0, b0, acc00);
            acc01 = WMMA16(a0, b1, acc01);
            acc10 = WMMA16(a1, b0, acc10);
            acc11 = WMMA16(a1, b1, acc11);
        }
        __syncthreads();
    }

    // --- store: C/D layout lane n = l16, VGPR r -> M = r + lh*8 ---
    const int gn0 = n0 + wn * 32 + l16, gn1 = gn0 + 16;
    #pragma unroll
    for (int r = 0; r < 8; ++r) {
        int gmA = m0 + wm * 32 + r + lh * 8;
        int gmB = gmA + 16;
        if (gmA < M) {
            float bv = bias ? bias[gmA] : 0.f;
            if (gn0 < N) { long i = (long)gmA * ldo + gn0; float v = acc00[r] + bv; if (accum) v += O[i]; O[i] = v; }
            if (gn1 < N) { long i = (long)gmA * ldo + gn1; float v = acc01[r] + bv; if (accum) v += O[i]; O[i] = v; }
        }
        if (gmB < M) {
            float bv = bias ? bias[gmB] : 0.f;
            if (gn0 < N) { long i = (long)gmB * ldo + gn0; float v = acc10[r] + bv; if (accum) v += O[i]; O[i] = v; }
            if (gn1 < N) { long i = (long)gmB * ldo + gn1; float v = acc11[r] + bv; if (accum) v += O[i]; O[i] = v; }
        }
    }
}

// ================== WMMA implicit-GEMM conv (KHxKW, pad) ===================
// NCHW, weights (Cout,Cin,KH,KW). Out[b] (Cout x HW) = sum_taps Wtap * Xshift.
// Same 64x128 tiling; K (input-channel) staged 64 deep per tap. Gather with
// spatial shift/zero-pad stays on the VALU path (not TDM-shaped).
__global__ __launch_bounds__(256)
void conv_wmma(const float* __restrict__ X, const float* __restrict__ Wt,
               float* __restrict__ O, int Cin, int Cout, int KH, int KW, int pad)
{
    const float* Xb = X + (long)blockIdx.z * Cin * HW_;
    float*       Ob = O + (long)blockIdx.z * Cout * HW_;
    const int m0 = blockIdx.y * 64, n0 = blockIdx.x * 128;

    __shared__ _Float16 As[64][72];
    __shared__ _Float16 Bs[64][136];

    const int t = threadIdx.x, lane = t & 31, wave = t >> 5;
    const int wm = wave >> 2, wn = wave & 3;
    const int lh = lane >> 4, l16 = lane & 15;
    const int taps = KH * KW;

    v8f acc00 = {}, acc01 = {}, acc10 = {}, acc11 = {};

    for (int tap = 0; tap < taps; ++tap) {
        const int ty = tap / KW - pad, tx = tap % KW - pad;
        for (int ci0 = 0; ci0 < Cin; ci0 += 64) {
            #pragma unroll
            for (int i = 0; i < 16; ++i) {      // A: weights for this tap (strided)
                int e = i * 256 + t, r = e >> 6, c = e & 63;
                float v = (m0 + r < Cout && ci0 + c < Cin)
                    ? Wt[((long)(m0 + r) * Cin + ci0 + c) * taps + tap] : 0.f;
                As[r][c] = (_Float16)v;
            }
            #pragma unroll
            for (int i = 0; i < 32; ++i) {      // B: shifted input, zero-padded
                int e = i * 256 + t, r = e >> 7, n = e & 127;
                int np = n0 + n;
                int oh = np >> 6, ow = np & 63;          // W_ == 64
                int ih = oh + ty, iw = ow + tx;
                float v = 0.f;
                if (ih >= 0 && ih < H_ && iw >= 0 && iw < W_ && (ci0 + r) < Cin)
                    v = Xb[(long)(ci0 + r) * HW_ + ih * W_ + iw];
                Bs[r][n] = (_Float16)v;
            }
            __syncthreads();

            #pragma unroll
            for (int kk = 0; kk < 64; kk += 32) {
                v16h a0, a1, b0, b1;
                #pragma unroll
                for (int j = 0; j < 8; ++j) {
                    a0[j]     = As[wm * 32 + l16][kk + lh * 8 + j];
                    a0[8 + j] = As[wm * 32 + l16][kk + 16 + lh * 8 + j];
                    a1[j]     = As[wm * 32 + 16 + l16][kk + lh * 8 + j];
                    a1[8 + j] = As[wm * 32 + 16 + l16][kk + 16 + lh * 8 + j];
                }
                #pragma unroll
                for (int j = 0; j < 16; ++j) {
                    int kl = kk + (j < 8 ? lh * 8 + j : 16 + lh * 8 + (j - 8));
                    b0[j] = Bs[kl][wn * 32 + l16];
                    b1[j] = Bs[kl][wn * 32 + 16 + l16];
                }
                acc00 = WMMA16(a0, b0, acc00);
                acc01 = WMMA16(a0, b1, acc01);
                acc10 = WMMA16(a1, b0, acc10);
                acc11 = WMMA16(a1, b1, acc11);
            }
            __syncthreads();
        }
    }

    const int gn0 = n0 + wn * 32 + l16, gn1 = gn0 + 16;
    #pragma unroll
    for (int r = 0; r < 8; ++r) {
        int gmA = m0 + wm * 32 + r + lh * 8;
        int gmB = gmA + 16;
        if (gmA < Cout) {
            Ob[(long)gmA * HW_ + gn0] = acc00[r];
            Ob[(long)gmA * HW_ + gn1] = acc01[r];
        }
        if (gmB < Cout) {
            Ob[(long)gmB * HW_ + gn0] = acc10[r];
            Ob[(long)gmB * HW_ + gn1] = acc11[r];
        }
    }
}

// =========================== reductions / pointwise ========================
__global__ void bn_stats_k(const float* __restrict__ x, float* __restrict__ mv,
                           int C, int HW, int Bn)
{
    const int c = blockIdx.x;
    const long n = (long)Bn * HW;
    float s = 0.f, s2 = 0.f;
    for (long i = threadIdx.x; i < n; i += blockDim.x) {
        long b = i / HW, p = i - b * HW;
        float v = x[(b * C + c) * (long)HW + p];
        s += v; s2 += v * v;
    }
    __shared__ float r1[256], r2[256];
    r1[threadIdx.x] = s; r2[threadIdx.x] = s2;
    __syncthreads();
    for (int o = 128; o > 0; o >>= 1) {
        if ((int)threadIdx.x < o) {
            r1[threadIdx.x] += r1[threadIdx.x + o];
            r2[threadIdx.x] += r2[threadIdx.x + o];
        }
        __syncthreads();
    }
    if (threadIdx.x == 0) {
        float m = r1[0] / (float)n;
        mv[c]     = m;
        mv[C + c] = r2[0] / (float)n - m * m;
    }
}

// act: 0 none, 1 relu, 2 silu. Optional residual add before activation.
__global__ void bn_apply_k(const float* __restrict__ x, float* __restrict__ y,
                           const float* __restrict__ mv,
                           const float* __restrict__ g, const float* __restrict__ bt,
                           const float* __restrict__ res,
                           int C, int HW, long total, float eps, int act)
{
    long i = (long)blockIdx.x * blockDim.x + threadIdx.x;
    if (i >= total) return;
    int c = (int)((i / HW) % C);
    float t = (x[i] - mv[c]) * rsqrtf(mv[C + c] + eps) * g[c] + bt[c];
    if (res) t += res[i];
    if (act == 1)      t = fmaxf(t, 0.f);
    else if (act == 2) t = t / (1.f + __expf(-t));
    y[i] = t;
}

__global__ void maxpool_k(const float* __restrict__ x, float* __restrict__ y,
                          long total)
{
    long i = (long)blockIdx.x * blockDim.x + threadIdx.x;
    if (i >= total) return;
    int p = (int)(i & (HW_ - 1));
    long base = i - p;
    int h = p >> 6, w = p & 63;
    float m = -3.4e38f;
    for (int dy = -1; dy <= 1; ++dy)
        for (int dx = -1; dx <= 1; ++dx) {
            int hh = h + dy, ww = w + dx;
            if (hh >= 0 && hh < H_ && ww >= 0 && ww < W_)
                m = fmaxf(m, x[base + hh * W_ + ww]);
        }
    y[i] = m;
}

__global__ void gn_stats_k(const float* __restrict__ x, float* __restrict__ mv,
                           long CHW, int Bn)
{
    const int b = blockIdx.x;
    const float* xb = x + (long)b * CHW;
    float s = 0.f, s2 = 0.f;
    for (long i = threadIdx.x; i < CHW; i += blockDim.x) {
        float v = xb[i]; s += v; s2 += v * v;
    }
    __shared__ float r1[256], r2[256];
    r1[threadIdx.x] = s; r2[threadIdx.x] = s2;
    __syncthreads();
    for (int o = 128; o > 0; o >>= 1) {
        if ((int)threadIdx.x < o) {
            r1[threadIdx.x] += r1[threadIdx.x + o];
            r2[threadIdx.x] += r2[threadIdx.x + o];
        }
        __syncthreads();
    }
    if (threadIdx.x == 0) {
        float m = r1[0] / (float)CHW;
        mv[b]      = m;
        mv[Bn + b] = r2[0] / (float)CHW - m * m;
    }
}

// es: optional per-channel extra scale (folds the depthwise 1x1 "conv")
__global__ void gn_apply_k(const float* __restrict__ x, float* __restrict__ y,
                           const float* __restrict__ mv,
                           const float* __restrict__ g, const float* __restrict__ bt,
                           const float* __restrict__ es,
                           int C, int HW, long total, int Bn, float eps)
{
    long i = (long)blockIdx.x * blockDim.x + threadIdx.x;
    if (i >= total) return;
    int b = (int)(i / ((long)C * HW));
    int c = (int)((i / HW) % C);
    float t = (x[i] - mv[b]) * rsqrtf(mv[Bn + b] + eps) * g[c] + bt[c];
    if (es) t *= es[c];
    y[i] = t;
}

__global__ void xsq_k(const float* __restrict__ e, float* __restrict__ xsq)
{
    long i = (long)blockIdx.x * blockDim.x + threadIdx.x;
    if (i >= (long)B_ * HW_) return;
    long b = i / HW_; int p = (int)(i % HW_);
    const float* eb = e + b * (long)C1_ * HW_ + p;
    float s = 0.f;
    for (int c = 0; c < C1_; ++c) { float v = eb[(long)c * HW_]; s += v * v; }
    xsq[i] = s;
}

__global__ void csq_k(const float* __restrict__ cw, float* __restrict__ csq)
{
    int k = threadIdx.x;
    if (k >= KC_) return;
    float s = 0.f;
    for (int c = 0; c < C1_; ++c) { float v = cw[k * C1_ + c]; s += v * v; }
    csq[k] = s;
}

// In-place: dot(K,N) per batch -> softmax over K of scale_k*(xsq-2dot+csq)
__global__ void softmax_k(float* __restrict__ aw, const float* __restrict__ xsq,
                          const float* __restrict__ csq, const float* __restrict__ scale)
{
    long i = (long)blockIdx.x * blockDim.x + threadIdx.x;
    if (i >= (long)B_ * HW_) return;
    long b = i / HW_; int p = (int)(i % HW_);
    float* awb = aw + b * (long)KC_ * HW_ + p;
    float l[KC_];
    float mx = -3.4e38f;
    const float xs = xsq[i];
    #pragma unroll
    for (int k = 0; k < KC_; ++k) {
        float d = xs - 2.f * awb[(long)k * HW_] + csq[k];
        l[k] = scale[k] * d;
        mx = fmaxf(mx, l[k]);
    }
    float s = 0.f;
    #pragma unroll
    for (int k = 0; k < KC_; ++k) { l[k] = __expf(l[k] - mx); s += l[k]; }
    float inv = 1.f / s;
    #pragma unroll
    for (int k = 0; k < KC_; ++k) awb[(long)k * HW_] = l[k] * inv;
}

__global__ void rowsum_k(const float* __restrict__ x, float* __restrict__ out, int n)
{
    const float* p = x + (long)blockIdx.x * n;
    float s = 0.f;
    for (int i = threadIdx.x; i < n; i += blockDim.x) s += p[i];
    __shared__ float r1[256];
    r1[threadIdx.x] = s;
    __syncthreads();
    for (int o = 128; o > 0; o >>= 1) {
        if ((int)threadIdx.x < o) r1[threadIdx.x] += r1[threadIdx.x + o];
        __syncthreads();
    }
    if (threadIdx.x == 0) out[blockIdx.x] = r1[0];
}

// enc_t[b][c][k] -= awsum[b][k] * cw[k][c]
__global__ void enc_adjust_k(float* __restrict__ enc, const float* __restrict__ awsum,
                             const float* __restrict__ cw)
{
    long i = (long)blockIdx.x * blockDim.x + threadIdx.x;
    if (i >= (long)B_ * C1_ * KC_) return;
    int k = (int)(i % KC_);
    long bc = i / KC_;
    int c = (int)(bc % C1_);
    long b = bc / C1_;
    enc[i] -= awsum[b * KC_ + k] * cw[(long)k * C1_ + c];
}

__global__ void bn1d_stats_k(const float* __restrict__ enc, float* __restrict__ mv)
{
    const int k = blockIdx.x;
    float s = 0.f, s2 = 0.f;
    for (int i = threadIdx.x; i < B_ * C1_; i += blockDim.x) {
        float v = enc[(long)i * KC_ + k];
        s += v; s2 += v * v;
    }
    __shared__ float r1[256], r2[256];
    r1[threadIdx.x] = s; r2[threadIdx.x] = s2;
    __syncthreads();
    for (int o = 128; o > 0; o >>= 1) {
        if ((int)threadIdx.x < o) {
            r1[threadIdx.x] += r1[threadIdx.x + o];
            r2[threadIdx.x] += r2[threadIdx.x + o];
        }
        __syncthreads();
    }
    if (threadIdx.x == 0) {
        const float n = (float)(B_ * C1_);
        float m = r1[0] / n;
        mv[k]       = m;
        mv[KC_ + k] = r2[0] / n - m * m;
    }
}

// en[b,c] = mean_k relu(bn1d(enc_t[b,c,k]))
__global__ void en_k(const float* __restrict__ enc, const float* __restrict__ mv,
                     const float* __restrict__ g, const float* __restrict__ bt,
                     float* __restrict__ en)
{
    int i = blockIdx.x * blockDim.x + threadIdx.x;
    if (i >= B_ * C1_) return;
    const float* row = enc + (long)i * KC_;
    float s = 0.f;
    for (int k = 0; k < KC_; ++k) {
        float t = (row[k] - mv[k]) * rsqrtf(mv[KC_ + k] + 1e-5f) * g[k] + bt[k];
        s += fmaxf(t, 0.f);
    }
    en[i] = s / (float)KC_;
}

__global__ void fc_sig_k(const float* __restrict__ en, const float* __restrict__ w,
                         const float* __restrict__ bias, float* __restrict__ gam)
{
    int i = blockIdx.x * blockDim.x + threadIdx.x;
    if (i >= B_ * C1_) return;
    int b = i / C1_, c = i % C1_;
    const float* er = en + (long)b * C1_;
    const float* wr = w + (long)c * C1_;
    float s = bias[c];
    for (int j = 0; j < C1_; ++j) s += er[j] * wr[j];
    gam[i] = 1.f / (1.f + __expf(-s));
}

// lvc_out = relu(x1 * (1 + gam[b,c]))
__global__ void gate_k(const float* __restrict__ x1, const float* __restrict__ gam,
                       float* __restrict__ o, long total)
{
    long i = (long)blockIdx.x * blockDim.x + threadIdx.x;
    if (i >= total) return;
    long bc = i / HW_;
    o[i] = fmaxf(x1[i] * (1.f + gam[bc]), 0.f);
}

// o = base + s[c] * add
__global__ void axpy_ch_k(const float* __restrict__ base, const float* __restrict__ s,
                          const float* __restrict__ add, float* __restrict__ o,
                          int C, int HW, long total)
{
    long i = (long)blockIdx.x * blockDim.x + threadIdx.x;
    if (i >= total) return;
    int c = (int)((i / HW) % C);
    o[i] = base[i] + s[c] * add[i];
}

__global__ void gelu_k(float* __restrict__ x, long total)
{
    long i = (long)blockIdx.x * blockDim.x + threadIdx.x;
    if (i >= total) return;
    float v = x[i];
    x[i] = 0.5f * v * (1.f + erff(v * 0.70710678118654752f));
}

// ================================ driver ===================================
extern "C" void kernel_launch(void* const* d_in, const int* in_sizes, int n_in,
                              void* d_out, int out_size, void* d_ws, size_t ws_size,
                              hipStream_t stream)
{
    const float* x        = (const float*)d_in[0];
    const float* stem_w   = (const float*)d_in[1];
    const float* stem_g   = (const float*)d_in[2];
    const float* stem_b   = (const float*)d_in[3];
    const float* cb_c1_w  = (const float*)d_in[4];
    const float* cb_bn1_g = (const float*)d_in[5];
    const float* cb_bn1_b = (const float*)d_in[6];
    const float* cb_c2_w  = (const float*)d_in[7];
    const float* cb_bn2_g = (const float*)d_in[8];
    const float* cb_bn2_b = (const float*)d_in[9];
    const float* cb_c3_w  = (const float*)d_in[10];
    const float* cb_bn3_g = (const float*)d_in[11];
    const float* cb_bn3_b = (const float*)d_in[12];
    const float* cb_res_w = (const float*)d_in[13];
    const float* resbn_g  = (const float*)d_in[14];
    const float* resbn_b  = (const float*)d_in[15];
    const float* lvc_c_w  = (const float*)d_in[16];
    const float* lvc_bn_g = (const float*)d_in[17];
    const float* lvc_bn_b = (const float*)d_in[18];
    const float* codewords= (const float*)d_in[19];
    const float* cw_scale = (const float*)d_in[20];
    const float* bn1d_g   = (const float*)d_in[21];
    const float* bn1d_b   = (const float*)d_in[22];
    const float* fc_w     = (const float*)d_in[23];
    const float* fc_b     = (const float*)d_in[24];
    const float* gn1_g    = (const float*)d_in[25];
    const float* gn1_b    = (const float*)d_in[26];
    const float* dw_w     = (const float*)d_in[27];
    const float* dw_bn_g  = (const float*)d_in[28];
    const float* dw_bn_b  = (const float*)d_in[29];
    const float* ls1      = (const float*)d_in[30];
    const float* gn2_g    = (const float*)d_in[31];
    const float* gn2_b    = (const float*)d_in[32];
    const float* fc1_w    = (const float*)d_in[33];
    const float* fc1_b    = (const float*)d_in[34];
    const float* fc2_w    = (const float*)d_in[35];
    const float* fc2_b    = (const float*)d_in[36];
    const float* ls2      = (const float*)d_in[37];
    const float* cnv1_w   = (const float*)d_in[38];
    const float* cnv1_b   = (const float*)d_in[39];
    float* out = (float*)d_out;
    (void)in_sizes; (void)n_in; (void)out_size; (void)ws_size;

    // ---- workspace layout (floats) ----
    float* ws = (float*)d_ws;
    const long S  = (long)B_ * C1_ * HW_;   // 8.4M
    const long SQ = (long)B_ * CB_ * HW_;   // 2.1M
    float* p_h   = ws;            // stem conv / later lvc_out
    float* p_hp  = p_h  + S;      // pooled stem (live through both branches)
    float* p_r   = p_hp + S;      // cb residual / later mlp_out
    float* p_x1  = p_r  + S;      // x1 / later h2
    float* p_e   = p_x1 + S;      // lvc 'e' / later u / z0
    float* p_t1  = p_e  + S;      // cb bottleneck
    float* p_t2  = p_t1 + SQ;
    float* p_aw  = p_t2 + SQ;     // dot / softmax weights (B,K,HW)
    float* p_zh  = p_aw + SQ;     // per-batch MLP hidden (HID x HW)
    float* p_z2  = p_zh + (long)HID_ * HW_;
    float* p_mv  = p_z2 + (long)C2_ * HW_;  // BN stats (2*HID max)
    float* p_gmv = p_mv  + 2 * HID_;
    float* p_xsq = p_gmv + 2 * B_;
    float* p_csq = p_xsq + (long)B_ * HW_;
    float* p_aws = p_csq + KC_;
    float* p_enc = p_aws + (long)B_ * KC_;  // enc_t (B,C,K)
    float* p_bmv = p_enc + (long)B_ * C1_ * KC_;
    float* p_en  = p_bmv + 2 * KC_;
    float* p_gam = p_en  + (long)B_ * C1_;

    const dim3 blk(256);
    auto g1 = [](long n) { return dim3((unsigned)((n + 255) / 256)); };
    const long CHW = (long)C1_ * HW_;
    const int NT = HW_ / 128;   // N tiles over spatial dim

    // ---- stem: 7x7 conv (WMMA implicit GEMM) + BN + ReLU + maxpool ----
    conv_wmma<<<dim3(NT, C1_/64, B_), blk, 0, stream>>>(x, stem_w, p_h, C1_, C1_, 7, 7, 3);
    bn_stats_k<<<C1_, blk, 0, stream>>>(p_h, p_mv, C1_, HW_, B_);
    bn_apply_k<<<g1(S), blk, 0, stream>>>(p_h, p_h, p_mv, stem_g, stem_b, nullptr, C1_, HW_, S, 1e-5f, 1);
    maxpool_k<<<g1(S), blk, 0, stream>>>(p_h, p_hp, S);

    // ---- ConvBlock: residual path ----
    gemm_wmma<0><<<dim3(NT, C2_/64, B_), blk, 0, stream>>>(
        cb_res_w, 0, C1_, p_hp, CHW, HW_, p_r, (long)C2_*HW_, HW_, nullptr, C2_, HW_, C1_, 0);
    bn_stats_k<<<C2_, blk, 0, stream>>>(p_r, p_mv, C2_, HW_, B_);
    bn_apply_k<<<g1(S), blk, 0, stream>>>(p_r, p_r, p_mv, resbn_g, resbn_b, nullptr, C2_, HW_, S, 1e-6f, 0);

    // ---- ConvBlock: bottleneck 1x1 -> 3x3 -> 1x1 ----
    gemm_wmma<0><<<dim3(NT, 1, B_), blk, 0, stream>>>(
        cb_c1_w, 0, C1_, p_hp, CHW, HW_, p_t1, (long)CB_*HW_, HW_, nullptr, CB_, HW_, C1_, 0);
    bn_stats_k<<<CB_, blk, 0, stream>>>(p_t1, p_mv, CB_, HW_, B_);
    bn_apply_k<<<g1(SQ), blk, 0, stream>>>(p_t1, p_t1, p_mv, cb_bn1_g, cb_bn1_b, nullptr, CB_, HW_, SQ, 1e-6f, 1);

    conv_wmma<<<dim3(NT, 1, B_), blk, 0, stream>>>(p_t1, cb_c2_w, p_t2, CB_, CB_, 3, 3, 1);
    bn_stats_k<<<CB_, blk, 0, stream>>>(p_t2, p_mv, CB_, HW_, B_);
    bn_apply_k<<<g1(SQ), blk, 0, stream>>>(p_t2, p_t2, p_mv, cb_bn2_g, cb_bn2_b, nullptr, CB_, HW_, SQ, 1e-6f, 1);

    gemm_wmma<0><<<dim3(NT, C2_/64, B_), blk, 0, stream>>>(
        cb_c3_w, 0, CB_, p_t2, (long)CB_*HW_, HW_, p_x1, (long)C2_*HW_, HW_, nullptr, C2_, HW_, CB_, 0);
    bn_stats_k<<<C2_, blk, 0, stream>>>(p_x1, p_mv, C2_, HW_, B_);
    bn_apply_k<<<g1(S), blk, 0, stream>>>(p_x1, p_x1, p_mv, cb_bn3_g, cb_bn3_b, p_r, C2_, HW_, S, 1e-6f, 1); // x1 = relu(bn3+r)

    // ---- LVC: 1x1 conv + BN + ReLU -> e ----
    gemm_wmma<0><<<dim3(NT, C1_/64, B_), blk, 0, stream>>>(
        lvc_c_w, 0, C1_, p_x1, CHW, HW_, p_e, CHW, HW_, nullptr, C1_, HW_, C1_, 0);
    bn_stats_k<<<C1_, blk, 0, stream>>>(p_e, p_mv, C1_, HW_, B_);
    bn_apply_k<<<g1(S), blk, 0, stream>>>(p_e, p_e, p_mv, lvc_bn_g, lvc_bn_b, nullptr, C1_, HW_, S, 1e-5f, 1);

    // ---- Encoding: dot GEMM (K x HW) = cw * e_b, softmax, aggregate GEMM ----
    xsq_k<<<g1((long)B_*HW_), blk, 0, stream>>>(p_e, p_xsq);
    csq_k<<<1, 64, 0, stream>>>(codewords, p_csq);
    gemm_wmma<0><<<dim3(NT, 1, B_), blk, 0, stream>>>(
        codewords, 0, C1_, p_e, CHW, HW_, p_aw, (long)KC_*HW_, HW_, nullptr, KC_, HW_, C1_, 0);
    softmax_k<<<g1((long)B_*HW_), blk, 0, stream>>>(p_aw, p_xsq, p_csq, cw_scale);
    rowsum_k<<<B_*KC_, blk, 0, stream>>>(p_aw, p_aws, HW_);
    // enc_t (C x K) per batch = e_b (C x HW) * aw_b^T   (transB, K-dim = HW)
    gemm_wmma<1><<<dim3(1, C1_/64, B_), blk, 0, stream>>>(
        p_e, CHW, HW_, p_aw, (long)KC_*HW_, HW_, p_enc, (long)C1_*KC_, KC_, nullptr, C1_, KC_, HW_, 0);
    enc_adjust_k<<<g1((long)B_*C1_*KC_), blk, 0, stream>>>(p_enc, p_aws, codewords);
    bn1d_stats_k<<<KC_, blk, 0, stream>>>(p_enc, p_bmv);
    en_k<<<g1(B_*C1_), blk, 0, stream>>>(p_enc, p_bmv, bn1d_g, bn1d_b, p_en);
    fc_sig_k<<<g1(B_*C1_), blk, 0, stream>>>(p_en, fc_w, fc_b, p_gam);
    gate_k<<<g1(S), blk, 0, stream>>>(p_x1, p_gam, p_h, S);   // lvc_out -> p_h

    // ---- LightMLP branch ----
    gn_stats_k<<<B_, blk, 0, stream>>>(p_hp, p_gmv, CHW, B_);
    gn_apply_k<<<g1(S), blk, 0, stream>>>(p_hp, p_e, p_gmv, gn1_g, gn1_b, dw_w, C1_, HW_, S, B_, 1e-5f); // u*dw
    bn_stats_k<<<C2_, blk, 0, stream>>>(p_e, p_mv, C2_, HW_, B_);
    bn_apply_k<<<g1(S), blk, 0, stream>>>(p_e, p_e, p_mv, dw_bn_g, dw_bn_b, nullptr, C2_, HW_, S, 1e-5f, 2); // silu
    axpy_ch_k<<<g1(S), blk, 0, stream>>>(p_hp, ls1, p_e, p_x1, C2_, HW_, S);  // h2 -> p_x1
    gn_stats_k<<<B_, blk, 0, stream>>>(p_x1, p_gmv, CHW, B_);
    gn_apply_k<<<g1(S), blk, 0, stream>>>(p_x1, p_e, p_gmv, gn2_g, gn2_b, nullptr, C1_, HW_, S, B_, 1e-5f); // z0 -> p_e

    for (int b = 0; b < B_; ++b) {   // per-batch MLP keeps hidden at 16 MB
        const float* z0b = p_e  + (long)b * CHW;
        const float* h2b = p_x1 + (long)b * CHW;
        float*      mlpb = p_r  + (long)b * CHW;
        gemm_wmma<0><<<dim3(NT, HID_/64, 1), blk, 0, stream>>>(
            fc1_w, 0, C1_, z0b, 0, HW_, p_zh, 0, HW_, fc1_b, HID_, HW_, C1_, 0);
        gelu_k<<<g1((long)HID_*HW_), blk, 0, stream>>>(p_zh, (long)HID_*HW_);
        gemm_wmma<0><<<dim3(NT, C2_/64, 1), blk, 0, stream>>>(
            fc2_w, 0, HID_, p_zh, 0, HW_, p_z2, 0, HW_, fc2_b, C2_, HW_, HID_, 0);
        axpy_ch_k<<<g1((long)C2_*HW_), blk, 0, stream>>>(h2b, ls2, p_z2, mlpb, C2_, HW_, (long)C2_*HW_);
    }

    // ---- fuse: cnv1 over concat == two GEMMs on split weight ----
    gemm_wmma<0><<<dim3(NT, C2_/64, B_), blk, 0, stream>>>(
        cnv1_w, 0, 2*C2_, p_h, CHW, HW_, out, (long)C2_*HW_, HW_, cnv1_b, C2_, HW_, C2_, 0);
    gemm_wmma<0><<<dim3(NT, C2_/64, B_), blk, 0, stream>>>(
        cnv1_w + C2_, 0, 2*C2_, p_r, CHW, HW_, out, (long)C2_*HW_, HW_, nullptr, C2_, HW_, C2_, 1);
}
